// ChronoSynth_88862873354949
// MI455X (gfx1250) — matrix operand order repaired and checked by reading, not compiled, not run on hardware
//
#include <hip/hip_runtime.h>

typedef _Float16 half_t;
typedef __attribute__((ext_vector_type(16))) _Float16 v16h;
typedef __attribute__((ext_vector_type(8)))  _Float16 v8h;
typedef __attribute__((ext_vector_type(8)))  float    v8f;

#define NFC 64
#define NIF 4
#define KS  5
#define K2T 25
#define HF  128
#define WF  128
#define HO  256
#define WO  256
#define HWF (HF*WF)
#define HWO (HO*WO)

__device__ __forceinline__ float lrelu_f(float v) { return v > 0.f ? v : 0.2f * v; }

// ---------------------------------------------------------------------------
// Implicit-GEMM 3x3 (pad 1) / 1x1 conv using v_wmma_f32_16x16x32_f16.
// Template-specialized so the K loop fully unrolls (back-to-back WMMA issue).
// Block = 128 threads (4 waves). Block tile: 32 couts x 64 pixels (row strip).
// Each wave: 16 couts x 32 pixels = 2 C tiles sharing one A fragment.
// A = packed weights f16 [CoutP][TAPS*CINP]; B = input patch staged in LDS.
// ---------------------------------------------------------------------------
template <int TAPS, int CINP>
__global__ __launch_bounds__(128) void conv_wmma_t(
    const half_t* __restrict__ in, const half_t* __restrict__ wgt,
    const float* __restrict__ bias, half_t* __restrict__ out,
    int CoutP, int CoValid, int H, int W, int relu,
    long inImgStride, long outImgStride)
{
    constexpr int KT   = TAPS * CINP;
    constexpr int ROWS = (TAPS == 9) ? 3 : 1;
    constexpr int COLS = (TAPS == 9) ? 66 : 64;
    __shared__ half_t ldsW[32 * KT];          // 32 couts x KT
    __shared__ half_t ldsI[ROWS * COLS * CINP];

    const int stripsPerRow = W >> 6;
    const int y        = blockIdx.x / stripsPerRow;
    const int xb       = (blockIdx.x % stripsPerRow) << 6;
    const int n        = blockIdx.z;
    const half_t* inN  = in + (long)n * inImgStride;

    // --- stage weight tile (32 couts x KT) ---
    {
        const int cb32 = blockIdx.y << 5;
        for (int idx = threadIdx.x; idx < 32 * KT; idx += 128)
            ldsW[idx] = wgt[(long)(cb32 + (idx / KT)) * KT + (idx % KT)];
    }

    // --- stage input patch ---
    if (TAPS == 9) {
        #pragma unroll 4
        for (int idx = threadIdx.x; idx < 3 * COLS * CINP; idx += 128) {
            int c = idx / (3 * COLS);
            int rem = idx % (3 * COLS);
            int r = rem / COLS, col = rem % COLS;
            int gy = y + r - 1, gx = xb + col - 1;
            half_t v = (half_t)0.f;
            if (gy >= 0 && gy < H && gx >= 0 && gx < W)
                v = inN[((long)c * H + gy) * W + gx];
            ldsI[(r * COLS + col) * CINP + c] = v;
        }
    } else {
        #pragma unroll 4
        for (int idx = threadIdx.x; idx < 64 * CINP; idx += 128) {
            int c = idx >> 6, col = idx & 63;
            ldsI[col * CINP + c] = inN[((long)c * H + y) * W + xb + col];
        }
    }
    __syncthreads();

    const int lane = threadIdx.x & 31;
    const int wave = threadIdx.x >> 5;
    const int ct   = wave & 1;            // cout half-tile
    const int ph   = wave >> 1;           // pixel half (32 px each)
    const int g    = lane >> 4;           // half-wave group
    const int ml   = lane & 15;           // cout row inside 16-tile (A)
    const int pl   = lane & 15;           // pixel inside 16-tile (B/C)
    const int coutBase = (blockIdx.y << 5) + (ct << 4);
    const int p0   = ph << 5;

    v8f acc0 = {}, acc1 = {};
    #pragma unroll
    for (int t = 0; t < TAPS; ++t) {
        const int dy  = (TAPS == 9) ? (t / 3) : 0;
        const int dx  = (TAPS == 9) ? (t % 3) : 0;
        const int col = p0 + pl + dx;
        const half_t* arow  = ldsW + (long)(ct * 16 + ml) * KT + t * CINP + (g << 4);
        const half_t* brow0 = ldsI + (long)(dy * COLS + col) * CINP + (g << 4);
        const half_t* brow1 = brow0 + 16 * CINP;   // +16 pixels
        #pragma unroll
        for (int cb = 0; cb < CINP; cb += 32) {
            // A fragment (16x32 f16): halves 0..15 -> K kb+16g..
            v8h alo = *(const v8h*)(arow + cb);
            v8h ahi = *(const v8h*)(arow + cb + 8);
            v16h a;
            #pragma unroll
            for (int q = 0; q < 8; ++q) { a[q] = alo[q]; a[q + 8] = ahi[q]; }
            // B fragments (32x16 f16): lane N=pixel, halves -> cin = cb + 16g + h
            v16h b0 = *(const v16h*)(brow0 + cb);
            v16h b1 = *(const v16h*)(brow1 + cb);
            acc0 = __builtin_amdgcn_wmma_f32_16x16x32_f16(
                false, a, false, b0, (short)0, acc0, false, false);
            acc1 = __builtin_amdgcn_wmma_f32_16x16x32_f16(
                false, a, false, b1, (short)0, acc1, false, false);
        }
    }

    // --- epilogue: bias + lrelu + mask padded couts, f16 store ---
    half_t* outN = out + (long)n * outImgStride;
    #pragma unroll
    for (int v = 0; v < 8; ++v) {
        int co = coutBase + v + (g << 3);     // C layout: M = v + 8g
        float v0 = acc0[v], v1 = acc1[v];
        if (co < CoValid) {
            float bv = bias[co];
            v0 += bv; v1 += bv;
            if (relu) { v0 = lrelu_f(v0); v1 = lrelu_f(v1); }
        } else {
            v0 = 0.f; v1 = 0.f;
        }
        half_t* orow = outN + ((long)co * H + y) * W + xb + p0 + pl;
        orow[0]  = (half_t)v0;
        orow[16] = (half_t)v1;
    }
}

// ---------------------------------------------------------------------------
// ConvTranspose2d k=3 s=2 p=1 op=1 (exact 2x upsample), direct VALU kernel.
// weight: fp32 [CoV][CiV][3][3] (reference uses it as conv weight on dilated lhs)
// ---------------------------------------------------------------------------
__global__ void convT_k(const half_t* __restrict__ in, const float* __restrict__ w3,
                        const float* __restrict__ b3, half_t* __restrict__ out,
                        int N, int CiV, int CinP, int CoValid, int CoutP, int h, int wd)
{
    const int H2 = 2 * h, W2 = 2 * wd;
    long idx = (long)blockIdx.x * blockDim.x + threadIdx.x;
    long total = (long)N * CoutP * H2 * W2;
    if (idx >= total) return;
    int X = (int)(idx % W2); long r = idx / W2;
    int Y = (int)(r % H2);   r /= H2;
    int co = (int)(r % CoutP);
    int n  = (int)(r / CoutP);
    if (co >= CoValid) { out[idx] = (half_t)0.f; return; }

    int yk[2], yi[2], ny = 0;
    int xk[2], xi[2], nx = 0;
    #pragma unroll
    for (int k = 0; k < 3; ++k) {
        int p = Y - 1 + k;
        if (p >= 0 && p <= 2 * h - 2 && (p & 1) == 0) { yk[ny] = k; yi[ny] = p >> 1; ny++; }
        p = X - 1 + k;
        if (p >= 0 && p <= 2 * wd - 2 && (p & 1) == 0) { xk[nx] = k; xi[nx] = p >> 1; nx++; }
    }
    float acc = b3[co];
    const half_t* inn = in + (long)n * CinP * h * wd;
    for (int ci = 0; ci < CiV; ++ci) {
        const float*  wc = w3 + ((long)co * CiV + ci) * 9;
        const half_t* ic = inn + (long)ci * h * wd;
        for (int a = 0; a < ny; ++a)
            for (int b = 0; b < nx; ++b)
                acc += wc[yk[a] * 3 + xk[b]] * (float)ic[yi[a] * wd + xi[b]];
    }
    out[idx] = (half_t)acc;
}

// --------------------------- elementwise helpers ---------------------------
__global__ void convert_feat(const float* __restrict__ f, half_t* __restrict__ o, long total)
{
    long idx = (long)blockIdx.x * blockDim.x + threadIdx.x;
    if (idx < total) o[idx] = (half_t)f[idx];
}

__global__ void build_occin(const float* __restrict__ feat, const float* __restrict__ timet,
                            half_t* __restrict__ o)
{
    long total = 2L * 288 * HWF;
    long idx = (long)blockIdx.x * blockDim.x + threadIdx.x;
    if (idx >= total) return;
    int p = (int)(idx % HWF); long r = idx / HWF;
    int j = (int)(r % 288);
    int b = (int)(r / 288);
    float v = 0.f;
    if (j < 260) {
        int i = j / 65, c = j % 65;
        if (c < NFC) {
            v = feat[(((long)b * NIF + i) * NFC + c) * HWF + p];
        } else {
            float t = timet[(long)b * HWF + p];
            v = (i == 0) ? t - 1.f : (i == 1) ? t : (i == 2) ? 1.f - t : 2.f - t;
        }
    }
    o[idx] = (half_t)v;
}

__global__ void pack_w(const float* __restrict__ w, half_t* __restrict__ pw,
                       int CoV, int CiV, int taps, int CinP, int CoutP)
{
    long total = (long)CoutP * taps * CinP;
    long idx = (long)blockIdx.x * blockDim.x + threadIdx.x;
    if (idx >= total) return;
    int ci = (int)(idx % CinP); long r = idx / CinP;
    int t  = (int)(r % taps);
    int co = (int)(r / taps);
    float v = 0.f;
    if (co < CoV && ci < CiV) v = w[((long)co * CiV + ci) * taps + t];
    pw[idx] = (half_t)v;
}

__global__ void softmax_occ(const half_t* __restrict__ in, float* __restrict__ out)
{
    long idx = (long)blockIdx.x * blockDim.x + threadIdx.x;
    if (idx >= 2L * HWO) return;
    int b = (int)(idx / HWO), p = (int)(idx % HWO);
    float v[NIF], mx = -1e30f;
    #pragma unroll
    for (int c = 0; c < NIF; ++c) {
        v[c] = (float)in[((long)b * 32 + c) * HWO + p];
        mx = fmaxf(mx, v[c]);
    }
    float s = 0.f;
    #pragma unroll
    for (int c = 0; c < NIF; ++c) { v[c] = __expf(v[c] - mx); s += v[c]; }
    float inv = 1.f / s;
    #pragma unroll
    for (int c = 0; c < NIF; ++c) out[((long)b * NIF + c) * HWO + p] = v[c] * inv;
}

__global__ void softmax_w25(const half_t* __restrict__ in, float* __restrict__ out)
{
    long idx = (long)blockIdx.x * blockDim.x + threadIdx.x;
    if (idx >= 2L * HWO) return;
    int b = (int)(idx / HWO), p = (int)(idx % HWO);
    float mx = -1e30f;
    for (int k = 0; k < K2T; ++k)
        mx = fmaxf(mx, (float)in[((long)b * 32 + k) * HWO + p]);
    float s = 0.f;
    for (int k = 0; k < K2T; ++k)
        s += __expf((float)in[((long)b * 32 + k) * HWO + p] - mx);
    float inv = 1.f / s;
    for (int k = 0; k < K2T; ++k)
        out[((long)b * K2T + k) * HWO + p] =
            __expf((float)in[((long)b * 32 + k) * HWO + p] - mx) * inv;
}

__global__ void zero_out_k(float* __restrict__ o, long total)
{
    long idx = (long)blockIdx.x * blockDim.x + threadIdx.x;
    if (idx < total) o[idx] = 0.f;
}

// ---------------------------------------------------------------------------
// AdaCoF synthesis for one frame i: deformable bilinear gather + 25-tap blend,
// occlusion-weighted accumulate into output (edge-replicate pad folded into
// clamp(y-2, 0, H-1)).
// ---------------------------------------------------------------------------
__global__ void synth_k(const float* __restrict__ frames, const float* __restrict__ occSm,
                        const float* __restrict__ wSm, const half_t* __restrict__ aOut,
                        const half_t* __restrict__ bOut, float* __restrict__ out, int i)
{
    long idx = (long)blockIdx.x * blockDim.x + threadIdx.x;
    if (idx >= 2L * HWO) return;
    int b = (int)(idx / HWO), p = (int)(idx % HWO);
    int Y = p / WO, X = p % WO;
    float occv = occSm[((long)b * NIF + i) * HWO + p];
    float acc[3] = {0.f, 0.f, 0.f};
    const float* fb = frames + ((long)i * 2 + b) * 3 * HWO;
    for (int k = 0; k < K2T; ++k) {
        float wk = wSm[((long)b * K2T + k) * HWO + p];
        float al = (float)aOut[((long)b * 32 + k) * HWO + p];
        float be = (float)bOut[((long)b * 32 + k) * HWO + p];
        float py = (float)Y + (float)(k / KS) + al;
        float px = (float)X + (float)(k % KS) + be;
        float y0f = floorf(py), x0f = floorf(px);
        float fy = py - y0f, fx = px - x0f;
        int y0 = min(max((int)y0f, 0), HO + 3);   // Hp-1 = 259
        int x0 = min(max((int)x0f, 0), WO + 3);
        int y1 = min(y0 + 1, HO + 3);
        int x1 = min(x0 + 1, WO + 3);
        int iy0 = min(max(y0 - 2, 0), HO - 1), iy1 = min(max(y1 - 2, 0), HO - 1);
        int ix0 = min(max(x0 - 2, 0), WO - 1), ix1 = min(max(x1 - 2, 0), WO - 1);
        float w00 = (1.f - fy) * (1.f - fx), w01 = (1.f - fy) * fx;
        float w10 = fy * (1.f - fx),         w11 = fy * fx;
        #pragma unroll
        for (int c = 0; c < 3; ++c) {
            const float* fc = fb + (long)c * HWO;
            float s = fc[iy0 * WO + ix0] * w00 + fc[iy0 * WO + ix1] * w01 +
                      fc[iy1 * WO + ix0] * w10 + fc[iy1 * WO + ix1] * w11;
            acc[c] += wk * s;
        }
    }
    #pragma unroll
    for (int c = 0; c < 3; ++c)
        out[((long)b * 3 + c) * HWO + p] += occv * acc[c];
}

// ---------------------------------------------------------------------------
extern "C" void kernel_launch(void* const* d_in, const int* in_sizes, int n_in,
                              void* d_out, int out_size, void* d_ws, size_t ws_size,
                              hipStream_t stream)
{
    (void)in_sizes; (void)n_in; (void)out_size; (void)ws_size;
    const float* features = (const float*)d_in[0];
    const float* frames   = (const float*)d_in[1];
    const float* timet    = (const float*)d_in[2];
    // nested params flattened sorted: alpha(3..10), beta(11..18), fuse_b(19),
    // fuse_w(20), occ(21..28), weight(29..36); within subnet: b1..b4, w1..w4
    auto P = [&](int i) { return (const float*)d_in[i]; };
    const int A_ = 3, B_ = 11, O_ = 21, Wt_ = 29;

    // workspace carve-out
    char* ws = (char*)d_ws;
    size_t off = 0;
    auto alloc = [&](size_t bytes) -> void* {
        void* p = ws + off;
        off += (bytes + 255) & ~(size_t)255;
        return p;
    };
    half_t* featF16 = (half_t*)alloc(8ULL * NFC * HWF * 2);
    half_t* occin   = (half_t*)alloc(2ULL * 288 * HWF * 2);
    half_t* xfuse   = (half_t*)alloc(2ULL * NFC * HWF * 2);
    half_t* t1      = (half_t*)alloc(2ULL * NFC * HWF * 2);
    half_t* t2      = (half_t*)alloc(2ULL * NFC * HWF * 2);
    half_t* tT      = (half_t*)alloc(2ULL * NFC * HWO * 2);
    half_t* occOut  = (half_t*)alloc(2ULL * 32 * HWO * 2);
    float*  occSm   = (float*) alloc(2ULL * NIF * HWO * 4);
    half_t* wOut    = (half_t*)alloc(2ULL * 32 * HWO * 2);
    half_t* aOut    = (half_t*)alloc(2ULL * 32 * HWO * 2);
    half_t* bOut    = (half_t*)alloc(2ULL * 32 * HWO * 2);
    float*  wSm     = (float*) alloc(2ULL * K2T * HWO * 4);
    half_t* pFuse   = (half_t*)alloc(64ULL * 288 * 2);
    half_t* pOw1    = (half_t*)alloc(64ULL * 576 * 2);
    half_t* pOw2    = (half_t*)alloc(64ULL * 576 * 2);
    half_t* pOw4    = (half_t*)alloc(32ULL * 576 * 2);
    half_t* pS1[3], *pS2[3], *pS4[3];
    for (int s = 0; s < 3; ++s) {
        pS1[s] = (half_t*)alloc(64ULL * 576 * 2);
        pS2[s] = (half_t*)alloc(32ULL * 576 * 2);
        pS4[s] = (half_t*)alloc(32ULL * 288 * 2);
    }

    auto blocks = [](long total) { return (unsigned)((total + 255) / 256); };

    // --- precision staging ---
    convert_feat<<<blocks(8L * NFC * HWF), 256, 0, stream>>>(features, featF16, 8L * NFC * HWF);
    build_occin<<<blocks(2L * 288 * HWF), 256, 0, stream>>>(features, timet, occin);

    // --- weight packing (zero-padded to WMMA-friendly shapes) ---
    auto pack = [&](const float* w, half_t* pw, int CoV, int CiV, int taps, int CinP, int CoutP) {
        pack_w<<<blocks((long)CoutP * taps * CinP), 256, 0, stream>>>(w, pw, CoV, CiV, taps, CinP, CoutP);
    };
    pack(P(20),     pFuse, 64, 260, 1, 288, 64);
    pack(P(O_ + 4), pOw1,  64, 64, 9, 64, 64);
    pack(P(O_ + 5), pOw2,  64, 64, 9, 64, 64);
    pack(P(O_ + 7), pOw4,  NIF, 64, 9, 64, 32);
    const int base_s[3] = {Wt_, A_, B_};
    for (int s = 0; s < 3; ++s) {
        pack(P(base_s[s] + 4), pS1[s], 64, 64, 9, 64, 64);
        pack(P(base_s[s] + 5), pS2[s], K2T, 64, 9, 64, 32);
        pack(P(base_s[s] + 7), pS4[s], K2T, K2T, 9, 32, 32);
    }

    auto launch_conv = [&](const half_t* in, const half_t* pw, const float* bias, half_t* out,
                           int CinP, int CoutP, int CoV, int H, int W, int taps, int relu,
                           long inStride, long outStride) {
        dim3 grid((unsigned)(H * (W / 64)), (unsigned)(CoutP / 32), 2);
        if (taps == 9 && CinP == 64)
            conv_wmma_t<9, 64><<<grid, 128, 0, stream>>>(in, pw, bias, out, CoutP, CoV,
                                                         H, W, relu, inStride, outStride);
        else if (taps == 9 && CinP == 32)
            conv_wmma_t<9, 32><<<grid, 128, 0, stream>>>(in, pw, bias, out, CoutP, CoV,
                                                         H, W, relu, inStride, outStride);
        else
            conv_wmma_t<1, 288><<<grid, 128, 0, stream>>>(in, pw, bias, out, CoutP, CoV,
                                                          H, W, relu, inStride, outStride);
    };

    // --- fuse 1x1 conv + lrelu ---
    launch_conv(occin, pFuse, P(19), xfuse, 288, 64, 64, HF, WF, 1, 1,
                288L * HWF, 64L * HWF);

    // --- occlusion subnet ---
    launch_conv(xfuse, pOw1, P(O_ + 0), t1, 64, 64, 64, HF, WF, 9, 1, 64L * HWF, 64L * HWF);
    launch_conv(t1,    pOw2, P(O_ + 1), t2, 64, 64, 64, HF, WF, 9, 1, 64L * HWF, 64L * HWF);
    convT_k<<<blocks(2L * 64 * HWO), 256, 0, stream>>>(t2, P(O_ + 6), P(O_ + 2), tT,
                                                       2, 64, 64, 64, 64, HF, WF);
    launch_conv(tT, pOw4, P(O_ + 3), occOut, 64, 32, NIF, HO, WO, 9, 0, 64L * HWO, 32L * HWO);
    softmax_occ<<<blocks(2L * HWO), 256, 0, stream>>>(occOut, occSm);

    zero_out_k<<<blocks(2L * 3 * HWO), 256, 0, stream>>>((float*)d_out, 2L * 3 * HWO);

    // --- per-frame weight/alpha/beta subnets + synthesis (keeps working set in L2) ---
    for (int i = 0; i < NIF; ++i) {
        const half_t* featI = featF16 + (long)i * NFC * HWF;   // image stride = 4*64*HWF
        auto chain = [&](int base, half_t* pw1, half_t* pw2, half_t* pw4, half_t* outb) {
            launch_conv(featI, pw1, P(base + 0), t1, 64, 64, 64, HF, WF, 9, 1,
                        4L * NFC * HWF, 64L * HWF);
            launch_conv(t1, pw2, P(base + 1), t2, 64, 32, K2T, HF, WF, 9, 1,
                        64L * HWF, 32L * HWF);
            convT_k<<<blocks(2L * 32 * HWO), 256, 0, stream>>>(t2, P(base + 6), P(base + 2), tT,
                                                               2, K2T, 32, K2T, 32, HF, WF);
            launch_conv(tT, pw4, P(base + 3), outb, 32, 32, K2T, HO, WO, 9, 0,
                        32L * HWO, 32L * HWO);
        };
        chain(Wt_, pS1[0], pS2[0], pS4[0], wOut);
        softmax_w25<<<blocks(2L * HWO), 256, 0, stream>>>(wOut, wSm);
        chain(A_,  pS1[1], pS2[1], pS4[1], aOut);
        chain(B_,  pS1[2], pS2[2], pS4[2], bOut);
        synth_k<<<blocks(2L * HWO), 256, 0, stream>>>(frames, occSm, wSm, aOut, bOut,
                                                      (float*)d_out, i);
    }
}
